// ClusterAttention_14620068675730
// MI455X (gfx1250) — matrix operand order repaired and checked by reading, not compiled, not run on hardware
//
#include <hip/hip_runtime.h>
#include <hip/hip_bf16.h>
#include <math.h>

// ---- problem constants (from reference) ----
#define B_    2
#define N_    4096
#define M_    32
#define C_    256
#define H_    8
#define CH_   32
#define T2_   625
#define BN_   (B_ * N_)          // 8192 rows
#define KDIM  256                // GEMM reduction dim (C)
#define QKV_COLS 768             // [Wq | Wkv] columns
#define SCALE_ 0.17677669529663687f   // CH^-0.5

typedef __attribute__((ext_vector_type(16))) _Float16 v16h;
typedef __attribute__((ext_vector_type(8)))  _Float16 v8h;
typedef __attribute__((ext_vector_type(8)))  float    v8f;

// -------------------- wave32 reductions --------------------
__device__ __forceinline__ float wave_max32(float v) {
  #pragma unroll
  for (int off = 16; off > 0; off >>= 1) v = fmaxf(v, __shfl_xor(v, off, 32));
  return v;
}
__device__ __forceinline__ float wave_sum32(float v) {
  #pragma unroll
  for (int off = 16; off > 0; off >>= 1) v += __shfl_xor(v, off, 32);
  return v;
}

// -------------------- prep kernels --------------------
__global__ void cvt_f32_to_f16(const float* __restrict__ in,
                               _Float16* __restrict__ out, int n) {
  int i = blockIdx.x * blockDim.x + threadIdx.x;
  if (i < n) out[i] = (_Float16)in[i];
}

// wt[col][k] = W[k][col] as f16; cols 0..255 from Wq(256x256), 256..767 from Wkv(256x512)
__global__ void pack_wqkv(const float* __restrict__ Wq, const float* __restrict__ Wkv,
                          _Float16* __restrict__ wt) {
  int i = blockIdx.x * blockDim.x + threadIdx.x;
  if (i >= QKV_COLS * KDIM) return;
  int col = i >> 8;
  int k   = i & 255;
  float v = (col < C_) ? Wq[k * C_ + col] : Wkv[k * (2 * C_) + (col - C_)];
  wt[i] = (_Float16)v;
}

__global__ void pack_wproj(const float* __restrict__ Wp, _Float16* __restrict__ wt) {
  int i = blockIdx.x * blockDim.x + threadIdx.x;
  if (i >= C_ * KDIM) return;
  int col = i >> 8;
  int k   = i & 255;
  wt[i] = (_Float16)Wp[k * C_ + col];
}

// pe_table[t][h] = pre_table[t][0..4] . W_pe[:,h] + b_pe[h]
__global__ void pe_table_kernel(const float* __restrict__ pre, const float* __restrict__ Wpe,
                                const float* __restrict__ bpe, float* __restrict__ pt) {
  int i = blockIdx.x * blockDim.x + threadIdx.x;
  if (i >= T2_ * H_) return;
  int t = i / H_, h = i % H_;
  float s = bpe[h];
  #pragma unroll
  for (int j = 0; j < 5; ++j) s += pre[t * 5 + j] * Wpe[j * H_ + h];
  pt[i] = s;
}

// -------------------- WMMA fragment loaders --------------------
// A: 16x32 f16 tile at (rowBase, kt) from row-major [*, KDIM]
__device__ __forceinline__ v16h load_a_frag(const _Float16* __restrict__ A,
                                            int rowBase, int kt, int lane) {
  int r    = lane & 15;
  int koff = (lane >> 4) * 8;   // lanes 16-31 hold K+8 block
  const _Float16* p = A + (size_t)(rowBase + r) * KDIM + kt + koff;
  union { v16h v; v8h h[2]; } a;
  a.h[0] = *(const v8h*)(p);        // K = koff .. koff+7
  a.h[1] = *(const v8h*)(p + 16);   // K = koff+16 .. koff+23
  return a.v;
}
// B: 32x16 f16 tile; Bt is column-major-of-W i.e. Bt[col][k], row-major [cols, KDIM]
__device__ __forceinline__ v16h load_b_frag(const _Float16* __restrict__ Bt,
                                            int colBase, int kt, int lane) {
  int c    = lane & 15;
  int koff = (lane >> 4) * 8;   // lanes 16-31 hold K+8 block
  const _Float16* p = Bt + (size_t)(colBase + c) * KDIM + kt + koff;
  union { v16h v; v8h h[2]; } b;
  b.h[0] = *(const v8h*)(p);        // K = koff .. koff+7
  b.h[1] = *(const v8h*)(p + 16);   // K = koff+16 .. koff+23
  return b.v;
}

// -------------------- QKV GEMM: (8192 x 256) x (256 x 768) --------------------
// One 16x64 output strip per wave (A fragment reused across 4 WMMAs per K-step),
// 4 waves/block. Fused bias + q/k/v split.
__global__ void gemm_qkv_wmma(const _Float16* __restrict__ A, const _Float16* __restrict__ Bt,
                              const float* __restrict__ bq, const float* __restrict__ bkv,
                              float* __restrict__ q, float* __restrict__ k,
                              float* __restrict__ v) {
  int wave = threadIdx.x >> 5;
  int lane = threadIdx.x & 31;
  int rowBase  = blockIdx.x * 16;
  int colStrip = (blockIdx.y * 4 + wave) * 64;   // 4 x 16-wide tiles

  v8f acc[4] = {{}, {}, {}, {}};
  #pragma unroll
  for (int kt = 0; kt < KDIM; kt += 32) {
    v16h a = load_a_frag(A, rowBase, kt, lane);
    #pragma unroll
    for (int t = 0; t < 4; ++t) {
      v16h b = load_b_frag(Bt, colStrip + t * 16, kt, lane);
      acc[t] = __builtin_amdgcn_wmma_f32_16x16x32_f16(false, a, false, b,
                                                      (short)0, acc[t], false, false);
    }
  }

  int mrow = (lane >> 4) * 8;
  #pragma unroll
  for (int t = 0; t < 4; ++t) {
    int colBase = colStrip + t * 16;
    int col = colBase + (lane & 15);
    float bias = (col < C_) ? bq[col] : bkv[col - C_];
    union { v8f v; float f[8]; } d; d.v = acc[t];
    #pragma unroll
    for (int rr = 0; rr < 8; ++rr) {
      int row = rowBase + mrow + rr;
      float val = d.f[rr] + bias;
      if (colBase < C_)          q[(size_t)row * C_ + col]          = val;
      else if (colBase < 2*C_)   k[(size_t)row * C_ + (col - C_)]   = val;
      else                       v[(size_t)row * C_ + (col - 2*C_)] = val;
    }
  }
}

// -------------------- output projection GEMM: (8192 x 256) x (256 x 256) --------------------
__global__ void gemm_proj_wmma(const _Float16* __restrict__ A, const _Float16* __restrict__ Bt,
                               const float* __restrict__ bias, float* __restrict__ out) {
  int wave = threadIdx.x >> 5;
  int lane = threadIdx.x & 31;
  int rowBase  = blockIdx.x * 16;
  int colStrip = (blockIdx.y * 4 + wave) * 64;

  v8f acc[4] = {{}, {}, {}, {}};
  #pragma unroll
  for (int kt = 0; kt < KDIM; kt += 32) {
    v16h a = load_a_frag(A, rowBase, kt, lane);
    #pragma unroll
    for (int t = 0; t < 4; ++t) {
      v16h b = load_b_frag(Bt, colStrip + t * 16, kt, lane);
      acc[t] = __builtin_amdgcn_wmma_f32_16x16x32_f16(false, a, false, b,
                                                      (short)0, acc[t], false, false);
    }
  }

  int mrow = (lane >> 4) * 8;
  #pragma unroll
  for (int t = 0; t < 4; ++t) {
    int col = colStrip + t * 16 + (lane & 15);
    float bv = bias[col];
    union { v8f v; float f[8]; } d; d.v = acc[t];
    #pragma unroll
    for (int rr = 0; rr < 8; ++rr) {
      int row = rowBase + mrow + rr;
      out[(size_t)row * C_ + col] = d.f[rr] + bv;
    }
  }
}

// -------------------- gathered cluster attention --------------------
// One wave per (b,n,h). Phase 1: lane = member (score). Phase 2: lane = channel (output).
// Writes f16 directly so the projection GEMM consumes it without a conversion pass.
__global__ void attn_kernel(const float* __restrict__ q, const float* __restrict__ k,
                            const float* __restrict__ v,
                            const int* __restrict__ member_idx,
                            const unsigned char* __restrict__ cmask,
                            const int* __restrict__ pe_idx,
                            const float* __restrict__ pe_table,
                            const float* __restrict__ blank_k,
                            const float* __restrict__ blank_v,
                            _Float16* __restrict__ out16) {
  int wid  = blockIdx.x * (blockDim.x >> 5) + (threadIdx.x >> 5); // [0, B*N*H)
  int lane = threadIdx.x & 31;
  int h  = wid & (H_ - 1);
  int bn = wid >> 3;            // b*N + n
  int b  = bn >> 12;            // / N_

  size_t mbase = (size_t)bn * M_ + lane;
  int mi   = member_idx[mbase];
  int msk  = cmask[mbase];
  int pidx = pe_idx[mbase];

  const float* qp = q + (size_t)bn * C_ + h * CH_;
  const float* kr = k + ((size_t)(b * N_ + mi)) * C_ + h * CH_;

  // score for member `lane`
  float s = 0.f;
  #pragma unroll
  for (int c = 0; c < CH_; ++c) s += qp[c] * kr[c];
  float score = s * SCALE_ + pe_table[pidx * H_ + h];
  if (!msk) score = -3.4028235e38f;

  // blank score (uniform across lanes)
  float sb = 0.f;
  #pragma unroll
  for (int c = 0; c < CH_; ++c) sb += qp[c] * blank_k[h * CH_ + c];
  float blank = sb * SCALE_;

  // softmax over 33 entries
  float mx = fmaxf(wave_max32(score), blank);
  float e  = __expf(score - mx);
  float eb = __expf(blank - mx);
  float denom = wave_sum32(e) + eb;
  float w  = e / denom;
  float wb = eb / denom;

  // output: lane = channel; broadcast member weight/index across lanes
  float accum = wb * blank_v[h * CH_ + lane];
  #pragma unroll
  for (int mm = 0; mm < M_; ++mm) {
    float wm  = __shfl(w, mm, 32);
    int   mim = __shfl(mi, mm, 32);
    const float* vr = v + ((size_t)(b * N_ + mim)) * C_ + h * CH_;
    accum += wm * vr[lane];
  }
  out16[(size_t)bn * C_ + h * CH_ + lane] = (_Float16)accum;
}

// -------------------- launcher --------------------
extern "C" void kernel_launch(void* const* d_in, const int* in_sizes, int n_in,
                              void* d_out, int out_size, void* d_ws, size_t ws_size,
                              hipStream_t stream) {
  const float* feat       = (const float*)d_in[0];
  const int*   member_idx = (const int*)d_in[1];
  const unsigned char* cmask = (const unsigned char*)d_in[2];
  const int*   pe_idx     = (const int*)d_in[3];
  // d_in[4] = global_attn (unused, always 0 in reference path)
  const float* pre_table  = (const float*)d_in[5];
  const float* Wq    = (const float*)d_in[6];
  const float* bq    = (const float*)d_in[7];
  const float* Wkv   = (const float*)d_in[8];
  const float* bkv   = (const float*)d_in[9];
  const float* Wpe   = (const float*)d_in[10];
  const float* bpe   = (const float*)d_in[11];
  const float* blank_k = (const float*)d_in[12];
  const float* blank_v = (const float*)d_in[13];
  const float* Wproj = (const float*)d_in[14];
  const float* bproj = (const float*)d_in[15];
  float* out = (float*)d_out;

  // workspace carve-out (256B aligned slices)
  char* ws = (char*)d_ws;
  size_t off = 0;
  auto take = [&](size_t bytes) -> char* {
    char* p = ws + off;
    off += (bytes + 255) & ~(size_t)255;
    return p;
  };
  _Float16* feat16  = (_Float16*)take((size_t)BN_ * C_ * sizeof(_Float16));
  _Float16* wqkvT   = (_Float16*)take((size_t)QKV_COLS * KDIM * sizeof(_Float16));
  _Float16* wprojT  = (_Float16*)take((size_t)C_ * KDIM * sizeof(_Float16));
  float*    peT     = (float*)take((size_t)T2_ * H_ * sizeof(float));
  float*    q_ws    = (float*)take((size_t)BN_ * C_ * sizeof(float));
  float*    k_ws    = (float*)take((size_t)BN_ * C_ * sizeof(float));
  float*    v_ws    = (float*)take((size_t)BN_ * C_ * sizeof(float));
  _Float16* aout16  = (_Float16*)take((size_t)BN_ * C_ * sizeof(_Float16));
  (void)in_sizes; (void)n_in; (void)out_size; (void)ws_size;

  // 1) prep
  {
    int n = BN_ * C_;
    cvt_f32_to_f16<<<(n + 255) / 256, 256, 0, stream>>>(feat, feat16, n);
    pack_wqkv<<<(QKV_COLS * KDIM + 255) / 256, 256, 0, stream>>>(Wq, Wkv, wqkvT);
    pack_wproj<<<(C_ * KDIM + 255) / 256, 256, 0, stream>>>(Wproj, wprojT);
    pe_table_kernel<<<(T2_ * H_ + 255) / 256, 256, 0, stream>>>(pre_table, Wpe, bpe, peT);
  }

  // 2) fused QKV GEMM (WMMA): rows 8192/16 = 512 blocks.x; cols 768/(4*64) = 3 blocks.y
  gemm_qkv_wmma<<<dim3(BN_ / 16, QKV_COLS / 256), 128, 0, stream>>>(
      feat16, wqkvT, bq, bkv, q_ws, k_ws, v_ws);

  // 3) gathered attention: B*N*H = 65536 waves, 8 waves/block; writes f16 directly
  attn_kernel<<<(B_ * N_ * H_) / 8, 256, 0, stream>>>(
      q_ws, k_ws, v_ws, member_idx, cmask, pe_idx, peT, blank_k, blank_v, aout16);

  // 4) output projection (WMMA) into d_out: cols 256/(4*64) = 1 block.y
  gemm_proj_wmma<<<dim3(BN_ / 16, C_ / 256), 128, 0, stream>>>(
      aout16, wprojT, bproj, out);
}
